// MGCD_15341623181677
// MI455X (gfx1250) — matrix operand order repaired and checked by reading, not compiled, not run on hardware
//
#include <hip/hip_runtime.h>
#include <math.h>

typedef __bf16 bf16;
typedef __attribute__((ext_vector_type(16))) __bf16 v16bf;
typedef __attribute__((ext_vector_type(8)))  float  v8f;
typedef __attribute__((ext_vector_type(4)))  int    iv4;

#define DEVI __device__ __forceinline__

// ---------------------------------------------------------------------------
// gfx1250 async global->LDS path (ASYNCcnt), guarded so compile never breaks.
// ---------------------------------------------------------------------------
#if defined(__has_builtin)
#  if __has_builtin(__builtin_amdgcn_global_load_async_to_lds_b128)
#    define HAS_ASYNC_LDS 1
#  endif
#endif
#ifndef HAS_ASYNC_LDS
#  define HAS_ASYNC_LDS 0
#endif

#if HAS_ASYNC_LDS
typedef __attribute__((address_space(1))) iv4 g_iv4;
typedef __attribute__((address_space(3))) iv4 l_iv4;
#endif

// 16-byte global->LDS copy (async when available).
DEVI void g2l16(const void* g, void* l) {
#if HAS_ASYNC_LDS
  __builtin_amdgcn_global_load_async_to_lds_b128(
      (g_iv4*)(uintptr_t)g, (l_iv4*)(uint32_t)(uintptr_t)l, 0, 0);
#else
  *(uint4*)l = *(const uint4*)g;
#endif
}

DEVI void g2l_commit() {
#if HAS_ASYNC_LDS
#  if __has_builtin(__builtin_amdgcn_s_wait_asynccnt)
  __builtin_amdgcn_s_wait_asynccnt(0);
#  else
  asm volatile("s_wait_asynccnt 0x0" ::: "memory");
#  endif
#endif
}

DEVI v8f wmma_bf16(v16bf a, v16bf b, v8f c) {
  // D = A(16x32 bf16) * B(32x16 bf16) + C(16x16 f32)
  return __builtin_amdgcn_wmma_f32_16x16x32_bf16(false, a, false, b, (short)0, c, false, false);
}

// A-fragment loader: rowptr = &tile[row][0] (row-major, K contiguous).
// lane kh = lane>>4 selects K halves {kh*8..+8} and {16+kh*8..+8}.
DEVI v16bf frag_a(const bf16* rowptr, int kh) {
  union { v16bf v; bf16 e[16]; } u;
#pragma unroll
  for (int i = 0; i < 8; ++i) {
    u.e[i]     = rowptr[kh * 8 + i];
    u.e[8 + i] = rowptr[16 + kh * 8 + i];
  }
  return u.v;
}

// B-fragment loader: colptr = &tile[col][0]; lane col = lane&15, K halves {kh*8..+8} and {16+kh*8..+8}.
DEVI v16bf frag_b(const bf16* colptr, int kh) {
  union { v16bf v; bf16 e[16]; } u;
#pragma unroll
  for (int i = 0; i < 8; ++i) {
    u.e[i]     = colptr[kh * 8 + i];
    u.e[8 + i] = colptr[16 + kh * 8 + i];
  }
  return u.v;
}

// ---------------------------------------------------------------------------
// 1) QKV projection for both tensors (X and MGCE_out).
//    Q,K stored bf16 as [tz][b][p][8]; V stored bf16 as [tz][b][c][p].
// ---------------------------------------------------------------------------
__global__ void qkv_kernel(const float* __restrict__ X, const float* __restrict__ M,
                           const float* __restrict__ wq, const float* __restrict__ bq,
                           const float* __restrict__ wk, const float* __restrict__ bk,
                           const float* __restrict__ wv, const float* __restrict__ bv,
                           bf16* __restrict__ Q, bf16* __restrict__ K, bf16* __restrict__ V) {
  const int p  = blockIdx.x * 256 + threadIdx.x;  // 0..1023
  const int b  = blockIdx.y;
  const int tz = blockIdx.z;
  const float* T = tz ? M : X;

  __shared__ float lwv[64 * 64];
  for (int i = threadIdx.x; i < 4096; i += 256) lwv[i] = wv[i];
  __syncthreads();

  float q[8] = {}, k[8] = {}, v[64] = {};
  const float* tcol = T + (size_t)b * 64 * 1024 + p;
  for (int c = 0; c < 64; ++c) {
    float x = tcol[(size_t)c * 1024];
#pragma unroll
    for (int i = 0; i < 8; ++i) { q[i] += wq[i * 64 + c] * x; k[i] += wk[i * 64 + c] * x; }
#pragma unroll
    for (int o = 0; o < 64; ++o) v[o] += lwv[o * 64 + c] * x;
  }
  bf16* Qp = Q + ((size_t)(tz * 32 + b) * 1024 + p) * 8;
  bf16* Kp = K + ((size_t)(tz * 32 + b) * 1024 + p) * 8;
  union { uint4 u; bf16 e[8]; } pq, pk;
#pragma unroll
  for (int i = 0; i < 8; ++i) { pq.e[i] = (bf16)(q[i] + bq[i]); pk.e[i] = (bf16)(k[i] + bk[i]); }
  *(uint4*)Qp = pq.u;
  *(uint4*)Kp = pk.u;
  bf16* Vp = V + ((size_t)(tz * 32 + b) * 64) * 1024 + p;
#pragma unroll
  for (int o = 0; o < 64; ++o) Vp[(size_t)o * 1024] = (bf16)(v[o] + bv[o]);
}

// ---------------------------------------------------------------------------
// 2) Flash-style channel self-attention, WMMA bf16.
//    Per block: one (tz, b, 64-row m-tile). Writes sa(t) = gamma*out + t.
// ---------------------------------------------------------------------------
__global__ void attn_kernel(const float* __restrict__ X, const float* __restrict__ M,
                            const bf16* __restrict__ Q, const bf16* __restrict__ K,
                            const bf16* __restrict__ V, const float* __restrict__ gamma_p,
                            float* __restrict__ saX, float* __restrict__ saM) {
  const int mt = blockIdx.x, b = blockIdx.y, tz = blockIdx.z;
  const float* T  = tz ? M : X;
  float*       SA = tz ? saM : saX;
  const bf16* Qb = Q + ((size_t)(tz * 32 + b) * 1024) * 8;
  const bf16* Kb = K + ((size_t)(tz * 32 + b) * 1024) * 8;
  const bf16* Vb = V + ((size_t)(tz * 32 + b) * 64) * 1024;

  const int tid = threadIdx.x, lane = tid & 31, w = tid >> 5;
  const int msub = w & 3, chalf = w >> 2;   // chalf doubles as nsub for the score phase
  const int m0 = mt * 64;

  __shared__ __align__(16) bf16  lQ[64][40], lK[32][40], lV[64][40], lP[64][40];
  __shared__ float lS[64][36];
  __shared__ float lmax[64], lsum[64], lalpha[64];

  // lQ: rows 0..63, data in cols 0..7, zero pad cols 8..31 (once).
  // lK pad region (cols 8..31) is also constant: zero it once here.
  {
    uint4 z = {};
    if (tid < 64) {
      g2l16(Qb + (size_t)(m0 + tid) * 8, &lQ[tid][0]);
      *(uint4*)&lQ[tid][8]  = z;
      *(uint4*)&lQ[tid][16] = z;
      *(uint4*)&lQ[tid][24] = z;
    } else if (tid < 160) {
      int idx = tid - 64, r = idx >> 2, ch = idx & 3;  // 24 rows x 3 chunks (+1 spare)
      if (ch < 3) *(uint4*)&lK[r][8 + ch * 8] = z;
    } else if (tid < 224) {
      lmax[tid - 160] = -1e30f;
      lsum[tid - 160] = 0.f;
    }
  }
  v8f acc0 = {}, acc1 = {};
  g2l_commit();
  __syncthreads();

  for (int n0 = 0; n0 < 1024; n0 += 32) {
    // stream K rows (16B each) and the full V tile (64 rows x 64B) into LDS
    if (tid < 32) g2l16(Kb + (size_t)(n0 + tid) * 8, &lK[tid][0]);
    {
      int r = tid >> 2, qc = tid & 3;
      g2l16(Vb + (size_t)r * 1024 + n0 + qc * 8, &lV[r][qc * 8]);
    }
    g2l_commit();
    __syncthreads();

    {   // S tile = Q^T K  (K-dim = 8, zero-padded to 32)
      v16bf qa = frag_a(&lQ[msub * 16 + (lane & 15)][0], lane >> 4);
      v16bf kf = frag_b(&lK[chalf * 16 + (lane & 15)][0], lane >> 4);
      v8f s = {};
      s = wmma_bf16(qa, kf, s);
#pragma unroll
      for (int i = 0; i < 8; ++i)
        lS[msub * 16 + ((lane >> 4) << 3) + i][chalf * 16 + (lane & 15)] = s[i];
    }
    __syncthreads();

    if (tid < 64) {  // online softmax over this 32-column block (one thread per row)
      int r = tid;
      float bm = -1e30f;
#pragma unroll
      for (int j = 0; j < 32; ++j) bm = fmaxf(bm, lS[r][j]);
      float nm = fmaxf(lmax[r], bm);
      float al = __expf(lmax[r] - nm);
      float rs = 0.f;
#pragma unroll
      for (int h = 0; h < 4; ++h) {
        union { uint4 u; bf16 e[8]; } pkk;
#pragma unroll
        for (int j = 0; j < 8; ++j) {
          float pp = __expf(lS[r][h * 8 + j] - nm);
          pkk.e[j] = (bf16)pp;
          rs += pp;
        }
        *(uint4*)&lP[r][h * 8] = pkk.u;
      }
      lalpha[r] = al; lmax[r] = nm; lsum[r] = lsum[r] * al + rs;
    }
    __syncthreads();

    {   // rescale and accumulate out += P * V^T
      v16bf pa = frag_a(&lP[msub * 16 + (lane & 15)][0], lane >> 4);
#pragma unroll
      for (int i = 0; i < 8; ++i) {
        float al = lalpha[msub * 16 + ((lane >> 4) << 3) + i];
        acc0[i] *= al; acc1[i] *= al;
      }
      v16bf v0 = frag_b(&lV[chalf * 32 + (lane & 15)][0], lane >> 4);
      v16bf v1 = frag_b(&lV[chalf * 32 + 16 + (lane & 15)][0], lane >> 4);
      acc0 = wmma_bf16(pa, v0, acc0);
      acc1 = wmma_bf16(pa, v1, acc1);
    }
    __syncthreads();
  }

  const float gamma = gamma_p[0];
#pragma unroll
  for (int i = 0; i < 8; ++i) {
    int mr = msub * 16 + ((lane >> 4) << 3) + i;
    float inv = 1.f / lsum[mr];
    int m = m0 + mr;
    int c0 = chalf * 32 + (lane & 15);
    size_t i0 = ((size_t)b * 64 + c0) * 1024 + m;
    size_t i1 = ((size_t)b * 64 + c0 + 16) * 1024 + m;
    SA[i0] = gamma * acc0[i] * inv + T[i0];
    SA[i1] = gamma * acc1[i] * inv + T[i1];
  }
}

// 3) Xa = sa(X) * sigmoid(sa(M)), in place over saX.
__global__ void combine_kernel(float* __restrict__ sx, const float* __restrict__ sm) {
  int idx = blockIdx.x * 256 + threadIdx.x;
  float a = sx[idx], g = sm[idx];
  sx[idx] = a * (1.f / (1.f + __expf(-g)));
}

// 4) Down transposed conv (k=(1,2), stride=(1,2)): out layout [b][o][n][w4].
__global__ void down_kernel(const float* __restrict__ Xa, const float* __restrict__ wgt,
                            const float* __restrict__ bias, float* __restrict__ out) {
  int idx = blockIdx.x * 256 + threadIdx.x;  // ((b*32+o)*512+n)*4+w
  int wo = idx & 3, n = (idx >> 2) & 511, o = (idx >> 11) & 31, b = idx >> 16;
  int win = wo >> 1, j = wo & 1;
  float acc = bias[o];
  const float* xp = Xa + (size_t)b * 64 * 1024 + n * 2 + win;
  for (int c = 0; c < 64; ++c) acc += xp[(size_t)c * 1024] * wgt[(c * 32 + o) * 2 + j];
  out[idx] = acc;
}

// 5) Training-mode BN stats over [32][32][2048]; out[c]=mean, out[32+c]=rsqrt(var+eps).
__global__ void stats_kernel(const float* __restrict__ src, float* __restrict__ out) {
  const int c = blockIdx.x, tid = threadIdx.x;
  __shared__ float s1[256], s2[256];
  float a = 0.f, sq = 0.f;
  for (int i = tid; i < 32 * 2048; i += 256) {
    int bb = i >> 11, s = i & 2047;
    float v = src[((size_t)(bb * 32 + c) << 11) + s];
    a += v; sq += v * v;
  }
  s1[tid] = a; s2[tid] = sq; __syncthreads();
  for (int st = 128; st > 0; st >>= 1) {
    if (tid < st) { s1[tid] += s1[tid + st]; s2[tid] += s2[tid + st]; }
    __syncthreads();
  }
  if (tid == 0) {
    const float cnt = 65536.f;
    float m = s1[0] / cnt;
    float var = s2[0] / cnt - m * m;
    out[c] = m; out[32 + c] = rsqrtf(var + 1e-5f);
  }
}

// 6) h' = relu(bn(h_down)) folded with fc1x weights, transposed to [x][b][col(c,g)][m] bf16.
__global__ void hprime_kernel(const float* __restrict__ hpre, const float* __restrict__ st,
                              const float* __restrict__ g, const float* __restrict__ bb,
                              const float* __restrict__ f1w, const float* __restrict__ f2w,
                              const float* __restrict__ f3w, bf16* __restrict__ hpT) {
  int idx = blockIdx.x * 256 + threadIdx.x;  // 32*32*512
  int m = idx & 511, c = (idx >> 9) & 31, b = idx >> 14;
  float mean = st[c], rstd = st[32 + c], gg = g[c], bv = bb[c];
  float h[4];
#pragma unroll
  for (int f = 0; f < 4; ++f) {
    float x = hpre[(((size_t)(b * 32 + c) * 512 + m) << 2) + f];
    h[f] = fmaxf((x - mean) * rstd * gg + bv, 0.f);
  }
  const float* fw[3] = { f1w, f2w, f3w };
#pragma unroll
  for (int x = 0; x < 3; ++x) {
#pragma unroll
    for (int gi = 0; gi < 4; ++gi) {
      float s = 0.f;
#pragma unroll
      for (int f = 0; f < 4; ++f) s += h[f] * fw[x][gi * 4 + f];
      hpT[((size_t)(x * 32 + b) * 128 + c * 4 + gi) * 512 + m] = (bf16)s;
    }
  }
}

// ---------------------------------------------------------------------------
// 7) Graph conv GEMM (flagship WMMA):  h2[b][n][col] = sum_x Adj_x[b] * h'_x[b]  + bias.
//    Per block: (n-tile of 64, batch). 8 waves -> 64x128 fp32 accumulators.
//    h' tiles stream memory->LDS via async b128; next adjacency slice prefetched.
// ---------------------------------------------------------------------------
__global__ void graph_gemm_kernel(const float* __restrict__ AC, const float* __restrict__ AD,
                                  const float* __restrict__ AW, const bf16* __restrict__ hpT,
                                  const float* __restrict__ b1, const float* __restrict__ b2,
                                  const float* __restrict__ b3, float* __restrict__ h2) {
  const int nt = blockIdx.x, b = blockIdx.y;
  const int tid = threadIdx.x, lane = tid & 31, w = tid >> 5;
  const int wn = w & 3, wc = w >> 2;
  const int n0 = nt * 64;

  __shared__ __align__(16) bf16 lA[64][40];   // adjacency tile: [n_local][m_local]
  __shared__ __align__(16) bf16 lB[128][40];  // h' tile:        [col][m_local]
  v8f acc[4] = {};

  const int ra = tid >> 2, sga = tid & 3;     // lA fill: 64 rows x 4 chunks of 8
  const float* adj[3] = { AC, AD, AW };
#pragma unroll 1
  for (int x = 0; x < 3; ++x) {
    const float* A  = adj[x] + ((size_t)b * 512 + n0) * 512;
    const bf16*  Bm = hpT + ((size_t)(x * 32 + b) * 128) * 512;
    for (int km = 0; km < 512; km += 32) {
      {  // adjacency fp32 -> bf16, 8 elements (32B in, 16B out) per thread
        const float* ap = A + (size_t)ra * 512 + km + sga * 8;
        float4 f0 = *(const float4*)ap, f1 = *(const float4*)(ap + 4);
        union { uint4 u; bf16 e[8]; } cv;
        cv.e[0] = (bf16)f0.x; cv.e[1] = (bf16)f0.y; cv.e[2] = (bf16)f0.z; cv.e[3] = (bf16)f0.w;
        cv.e[4] = (bf16)f1.x; cv.e[5] = (bf16)f1.y; cv.e[6] = (bf16)f1.z; cv.e[7] = (bf16)f1.w;
        *(uint4*)&lA[ra][sga * 8] = cv.u;
        if (km + 32 < 512) __builtin_prefetch(ap + 32, 0, 1);  // next adjacency k-slice
      }
#pragma unroll
      for (int cc = 0; cc < 2; ++cc) {  // h' tile: 128 rows x 4 chunks of 16B, 2 per thread
        int ci = tid + cc * 256;
        int r = ci >> 2, qc = ci & 3;
        g2l16(Bm + (size_t)r * 512 + km + qc * 8, &lB[r][qc * 8]);
      }
      g2l_commit();
      __syncthreads();
      v16bf af = frag_a(&lA[wn * 16 + (lane & 15)][0], lane >> 4);
#pragma unroll
      for (int j = 0; j < 4; ++j) {
        v16bf bfr = frag_b(&lB[wc * 64 + j * 16 + (lane & 15)][0], lane >> 4);
        acc[j] = wmma_bf16(af, bfr, acc[j]);
      }
      __syncthreads();
    }
  }
#pragma unroll
  for (int j = 0; j < 4; ++j) {
    int col = wc * 64 + j * 16 + (lane & 15);
    int gi = col & 3;
    float bias = b1[gi] + b2[gi] + b3[gi];
#pragma unroll
    for (int i = 0; i < 8; ++i) {
      int n = n0 + wn * 16 + ((lane >> 4) << 3) + i;
      h2[((size_t)b * 512 + n) * 128 + col] = acc[j][i] + bias;
    }
  }
}

// 8) Up transposed conv k=(1,3), pad=(0,1): input [b][n][c*4+w], output [b][o][n][t].
__global__ void up_kernel(const float* __restrict__ h2, const float* __restrict__ wgt,
                          const float* __restrict__ bias, float* __restrict__ out) {
  const int b = blockIdx.y, n0 = blockIdx.x * 8;
  const int tid = threadIdx.x, gi = tid >> 5, o = tid & 31;
  __shared__ __align__(16) float lrow[8][128];
  __shared__ float lw[3072];
  for (int i = tid; i < 3072; i += 256) lw[i] = wgt[i];
  {
    int r = tid >> 5, qc = tid & 31;  // 8 rows x 32 chunks of 4 floats
    *(float4*)&lrow[r][qc * 4] = *(const float4*)(h2 + ((size_t)b * 512 + n0 + r) * 128 + qc * 4);
  }
  __syncthreads();
  float y[4]; float bo = bias[o];
#pragma unroll
  for (int t = 0; t < 4; ++t) y[t] = bo;
  for (int c = 0; c < 32; ++c) {
#pragma unroll
    for (int k = 0; k < 3; ++k) {
      float wv = lw[(c * 32 + o) * 3 + k];
#pragma unroll
      for (int t = 0; t < 4; ++t) {
        int i = t + 1 - k;
        if (i >= 0 && i < 4) y[t] += lrow[gi][c * 4 + i] * wv;
      }
    }
  }
  size_t base = ((size_t)(b * 32 + o) * 512 + n0 + gi) * 4;
#pragma unroll
  for (int t = 0; t < 4; ++t) out[base + t] = y[t];
}

// 9) Residual transposed conv k=(1,2), dilation=(1,2) on Xa: output [b][o][n][t], t = win + 2k.
__global__ void res_kernel(const float* __restrict__ Xa, const float* __restrict__ wgt,
                           const float* __restrict__ bias, float* __restrict__ out) {
  const int b = blockIdx.y, n0 = blockIdx.x * 8;
  const int tid = threadIdx.x, gi = tid >> 5, o = tid & 31;
  __shared__ float lrow[8][128];  // [r][c*2+win]
  __shared__ float lw[4096];
  for (int i = tid; i < 4096; i += 256) lw[i] = wgt[i];
  for (int e = tid; e < 1024; e += 256) {
    int r = e >> 7, cj = e & 127;
    int c = cj >> 1, j = cj & 1;
    lrow[r][cj] = Xa[((size_t)b * 64 + c) * 1024 + (n0 + r) * 2 + j];
  }
  __syncthreads();
  float y[4]; float bo = bias[o];
#pragma unroll
  for (int t = 0; t < 4; ++t) y[t] = bo;
  for (int c = 0; c < 64; ++c) {
    float x0 = lrow[gi][c * 2], x1 = lrow[gi][c * 2 + 1];
    float w0 = lw[(c * 32 + o) * 2], w1 = lw[(c * 32 + o) * 2 + 1];
    y[0] += x0 * w0; y[1] += x1 * w0; y[2] += x0 * w1; y[3] += x1 * w1;
  }
  size_t base = ((size_t)(b * 32 + o) * 512 + n0 + gi) * 4;
#pragma unroll
  for (int t = 0; t < 4; ++t) out[base + t] = y[t];
}

// 10) h3 = relu(relu(bn_up(up)) + relu(bn_res(res))), in place over up buffer.
__global__ void fuse_kernel(float* __restrict__ up, const float* __restrict__ res,
                            const float* __restrict__ stU, const float* __restrict__ stR,
                            const float* __restrict__ gu, const float* __restrict__ bu,
                            const float* __restrict__ gr, const float* __restrict__ br) {
  int idx = blockIdx.x * 256 + threadIdx.x;
  int o = (idx >> 11) & 31;
  float u = fmaxf((up[idx] - stU[o]) * stU[32 + o] * gu[o] + bu[o], 0.f);
  float r = fmaxf((res[idx] - stR[o]) * stR[32 + o] * gr[o] + br[o], 0.f);
  up[idx] = fmaxf(u + r, 0.f);
}

// 11) Up1 transposed conv (same geometry as up), input layout [b][c][n][t].
__global__ void up1_kernel(const float* __restrict__ h3, const float* __restrict__ wgt,
                           const float* __restrict__ bias, float* __restrict__ out) {
  const int b = blockIdx.y, n0 = blockIdx.x * 8;
  const int tid = threadIdx.x, gi = tid >> 5, o = tid & 31;
  __shared__ __align__(16) float lrow[8][128];  // [r][c*4+t]
  __shared__ float lw[3072];
  for (int i = tid; i < 3072; i += 256) lw[i] = wgt[i];
  {
    int r = tid >> 5, cc = tid & 31;  // 8 rows x 32 channels; 16B per (r,c)
    *(float4*)&lrow[r][cc * 4] = *(const float4*)(h3 + ((size_t)(b * 32 + cc) * 512 + n0 + r) * 4);
  }
  __syncthreads();
  float y[4]; float bo = bias[o];
#pragma unroll
  for (int t = 0; t < 4; ++t) y[t] = bo;
  for (int c = 0; c < 32; ++c) {
#pragma unroll
    for (int k = 0; k < 3; ++k) {
      float wv = lw[(c * 32 + o) * 3 + k];
#pragma unroll
      for (int t = 0; t < 4; ++t) {
        int i = t + 1 - k;
        if (i >= 0 && i < 4) y[t] += lrow[gi][c * 4 + i] * wv;
      }
    }
  }
  size_t base = ((size_t)(b * 32 + o) * 512 + n0 + gi) * 4;
#pragma unroll
  for (int t = 0; t < 4; ++t) out[base + t] = y[t];
}

// 12) Final BN + ReLU -> d_out (layout [B][Cout][N][4]).
__global__ void final_kernel(const float* __restrict__ src, const float* __restrict__ st,
                             const float* __restrict__ g, const float* __restrict__ bb,
                             float* __restrict__ out) {
  int idx = blockIdx.x * 256 + threadIdx.x;
  int o = (idx >> 11) & 31;
  out[idx] = fmaxf((src[idx] - st[o]) * st[32 + o] * g[o] + bb[o], 0.f);
}

extern "C" void kernel_launch(void* const* d_in, const int* in_sizes, int n_in,
                              void* d_out, int out_size, void* d_ws, size_t ws_size,
                              hipStream_t stream) {
  (void)in_sizes; (void)n_in; (void)out_size; (void)ws_size;
  const float* M    = (const float*)d_in[0];
  const float* X    = (const float*)d_in[1];
  const float* AC   = (const float*)d_in[2];
  const float* AD   = (const float*)d_in[3];
  const float* AW   = (const float*)d_in[4];
  const float* wq   = (const float*)d_in[5];
  const float* bq   = (const float*)d_in[6];
  const float* wk   = (const float*)d_in[7];
  const float* bk   = (const float*)d_in[8];
  const float* wv   = (const float*)d_in[9];
  const float* bv   = (const float*)d_in[10];
  const float* gam  = (const float*)d_in[11];
  const float* dw   = (const float*)d_in[12];
  const float* db   = (const float*)d_in[13];
  const float* bng  = (const float*)d_in[14];
  const float* bnb  = (const float*)d_in[15];
  const float* f1w  = (const float*)d_in[16];
  const float* f1b  = (const float*)d_in[17];
  const float* f2w  = (const float*)d_in[18];
  const float* f2b  = (const float*)d_in[19];
  const float* f3w  = (const float*)d_in[20];
  const float* f3b  = (const float*)d_in[21];
  const float* upw  = (const float*)d_in[22];
  const float* upb  = (const float*)d_in[23];
  const float* bug  = (const float*)d_in[24];
  const float* bub  = (const float*)d_in[25];
  const float* rw   = (const float*)d_in[26];
  const float* rb   = (const float*)d_in[27];
  const float* brg  = (const float*)d_in[28];
  const float* brb  = (const float*)d_in[29];
  const float* u1w  = (const float*)d_in[30];
  const float* u1b  = (const float*)d_in[31];
  const float* b1g  = (const float*)d_in[32];
  const float* b1b  = (const float*)d_in[33];

  char* ws = (char*)d_ws;
  const size_t MB = 1024 * 1024;
  float* saX    = (float*)(ws);            // 8 MB  [b][c][p]  (becomes Xa)
  float* saM    = (float*)(ws + 8  * MB);  // 8 MB
  bf16*  Q      = (bf16*) (ws + 16 * MB);  // 1 MB  [tz][b][p][8]
  bf16*  Kq     = (bf16*) (ws + 17 * MB);  // 1 MB
  bf16*  V      = (bf16*) (ws + 18 * MB);  // 8 MB  [tz][b][c][p]
  float* hpre   = (float*)(ws + 26 * MB);  // 8 MB  [b][c32][n][4]
  bf16*  hpT    = (bf16*) (ws + 34 * MB);  // 12 MB [x][b][col128][m512]
  float* h2     = (float*)(ws + 46 * MB);  // 8 MB  [b][n][128]
  float* uppre  = (float*)(ws + 54 * MB);  // 8 MB  [b][o][n][4]
  float* respre = (float*)(ws + 62 * MB);  // 8 MB
  float* up1pre = (float*)(ws + 70 * MB);  // 8 MB
  float* stats  = (float*)(ws + 78 * MB);  // 4 stages x 64 floats

  qkv_kernel<<<dim3(4, 32, 2), 256, 0, stream>>>(X, M, wq, bq, wk, bk, wv, bv, Q, Kq, V);
  attn_kernel<<<dim3(16, 32, 2), 256, 0, stream>>>(X, M, Q, Kq, V, gam, saX, saM);
  combine_kernel<<<8192, 256, 0, stream>>>(saX, saM);
  down_kernel<<<8192, 256, 0, stream>>>(saX, dw, db, hpre);
  stats_kernel<<<32, 256, 0, stream>>>(hpre, stats + 0);
  hprime_kernel<<<2048, 256, 0, stream>>>(hpre, stats + 0, bng, bnb, f1w, f2w, f3w, hpT);
  graph_gemm_kernel<<<dim3(8, 32), 256, 0, stream>>>(AC, AD, AW, hpT, f1b, f2b, f3b, h2);
  up_kernel<<<dim3(64, 32), 256, 0, stream>>>(h2, upw, upb, uppre);
  res_kernel<<<dim3(64, 32), 256, 0, stream>>>(saX, rw, rb, respre);
  stats_kernel<<<32, 256, 0, stream>>>(uppre, stats + 64);
  stats_kernel<<<32, 256, 0, stream>>>(respre, stats + 128);
  fuse_kernel<<<8192, 256, 0, stream>>>(uppre, respre, stats + 64, stats + 128, bug, bub, brg, brb);
  up1_kernel<<<dim3(64, 32), 256, 0, stream>>>(uppre, u1w, u1b, up1pre);
  stats_kernel<<<32, 256, 0, stream>>>(up1pre, stats + 192);
  final_kernel<<<8192, 256, 0, stream>>>(up1pre, stats + 192, b1g, b1b, (float*)d_out);
}
